// MultiHeadAttention_18897856102411
// MI455X (gfx1250) — compile-verified, matrix operand order checked by reading, not run on hardware
//
#include <hip/hip_runtime.h>
#include <hip/hip_bf16.h>
#include <math.h>

#define BATCH 2
#define SEQ   2048
#define DMODEL 1024
#define NHEAD 16
#define DHEAD 64

typedef __attribute__((ext_vector_type(16))) _Float16 v16h;
typedef __attribute__((ext_vector_type(8)))  _Float16 v8h;
typedef __attribute__((ext_vector_type(4)))  _Float16 v4h;
typedef __attribute__((ext_vector_type(8)))  float    v8f;
typedef __attribute__((ext_vector_type(4)))  float    v4f;

__device__ __forceinline__ v8f wmma_f16(v16h a, v16h b, v8f c) {
  // D = A(16x32 f16) x B(32x16 f16) + C(16x16 f32)
  return __builtin_amdgcn_wmma_f32_16x16x32_f16(false, a, false, b, (short)0, c,
                                                false, false);
}

// Fragment layout (ISA 7.12.2, 16-bit 16x32): lane holds two contiguous
// 8-element K-chunks at k = kk and k = kk+16, kk = 8*(lane>=16).
__device__ __forceinline__ v16h ld_frag(const _Float16* p0, const _Float16* p1) {
  union { v16h v; v8h h[2]; } u;
  u.h[0] = *(const v8h*)p0;
  u.h[1] = *(const v8h*)p1;
  return u.v;
}

// DPP16 cross-lane move (stays within a 16-lane row; wave32, full EXEC).
template <int CTRL>
__device__ __forceinline__ float dpp_mov(float x) {
  int i = __builtin_bit_cast(int, x);
  int r = __builtin_amdgcn_update_dpp(i, i, CTRL, 0xf, 0xf, true);
  return __builtin_bit_cast(float, r);
}
// 16-lane butterfly: xor1 (quad_perm 1,0,3,2), xor2 (quad_perm 2,3,0,1),
// xor7 (row_half_mirror), xor15 (row_mirror).
__device__ __forceinline__ float redmax16(float x) {
  x = fmaxf(x, dpp_mov<0x0B1>(x));
  x = fmaxf(x, dpp_mov<0x04E>(x));
  x = fmaxf(x, dpp_mov<0x141>(x));
  x = fmaxf(x, dpp_mov<0x140>(x));
  return x;
}
__device__ __forceinline__ float redsum16(float x) {
  x += dpp_mov<0x0B1>(x);
  x += dpp_mov<0x04E>(x);
  x += dpp_mov<0x141>(x);
  x += dpp_mov<0x140>(x);
  return x;
}

__device__ __forceinline__ v4h ld4_cvt(const float* p) {
  v4f a = *(const v4f*)p;
  v4h r;
  r.x = (_Float16)a.x; r.y = (_Float16)a.y;
  r.z = (_Float16)a.z; r.w = (_Float16)a.w;
  return r;
}
__device__ __forceinline__ v4h ld4_cvt(const _Float16* p) {
  return *(const v4h*)p;
}

// C = A[M,K] * W[N,K]^T  (torch Linear), f16 WMMA, f32 accumulate.
// MODE 0: store f16 to [B,H,S,DH]   (Q, K)
// MODE 1: store f16 to [B,H,DH,S]   (V transposed)
// MODE 2: store f32 to [M,N] + bias (output projection -> d_out)
template <typename AT, int MODE>
__global__ void __launch_bounds__(256)
gemm_wmma(const AT* __restrict__ A, const float* __restrict__ W,
          const float* __restrict__ bias, void* __restrict__ OutP)
{
  const int N = DMODEL, K = DMODEL;
  __shared__ __align__(16) _Float16 As[128][40];  // +8 pad vs 64 banks
  __shared__ __align__(16) _Float16 Bs[64][40];
  const int m0  = blockIdx.y * 128;
  const int n0  = blockIdx.x * 64;
  const int tid = threadIdx.x;
  const int wid = tid >> 5, lane = tid & 31;
  const int wm = wid >> 1, wn = wid & 1;          // 4x2 wave grid, 32x32 each
  const int hl = lane >> 4, nl = lane & 15, kk = hl * 8;

  v8f acc[2][2] = {};
  for (int k0 = 0; k0 < K; k0 += 32) {
#pragma unroll
    for (int i = 0; i < 4; ++i) {                 // A tile: 128x32, vec4/thread
      int idx = (tid + i * 256) * 4;
      int r = idx >> 5, c = idx & 31;
      *(v4h*)&As[r][c] = ld4_cvt(&A[(size_t)(m0 + r) * K + (k0 + c)]);
    }
#pragma unroll
    for (int i = 0; i < 2; ++i) {                 // B tile: 64x32
      int idx = (tid + i * 256) * 4;
      int r = idx >> 5, c = idx & 31;
      *(v4h*)&Bs[r][c] = ld4_cvt(&W[(size_t)(n0 + r) * K + (k0 + c)]);
    }
    if (k0 + 32 < K)   // warm L2 for next K-slab (global_prefetch_b8)
      __builtin_prefetch(&A[(size_t)(m0 + (tid >> 1)) * K + (k0 + 32)], 0, 1);
    __syncthreads();

    v16h af[2], bf[2];
#pragma unroll
    for (int fm = 0; fm < 2; ++fm) {
      const _Float16* p = &As[wm * 32 + fm * 16 + nl][kk];
      af[fm] = ld_frag(p, p + 16);
    }
#pragma unroll
    for (int fn = 0; fn < 2; ++fn) {
      const _Float16* p = &Bs[wn * 32 + fn * 16 + nl][kk];
      bf[fn] = ld_frag(p, p + 16);
    }
#pragma unroll
    for (int fm = 0; fm < 2; ++fm)
#pragma unroll
      for (int fn = 0; fn < 2; ++fn)
        acc[fm][fn] = wmma_f16(af[fm], bf[fn], acc[fm][fn]);
    __syncthreads();
  }

  // C layout: vgpr r, lane l -> m = r + 8*(l>>4), n = l&15
#pragma unroll
  for (int fm = 0; fm < 2; ++fm)
#pragma unroll
    for (int fn = 0; fn < 2; ++fn)
#pragma unroll
      for (int r = 0; r < 8; ++r) {
        int m = m0 + wm * 32 + fm * 16 + r + 8 * hl;
        int e = n0 + wn * 32 + fn * 16 + nl;
        float v = acc[fm][fn][r];
        if (MODE == 2) {
          ((float*)OutP)[(size_t)m * N + e] = v + bias[e];
        } else {
          int b = m / SEQ, s = m % SEQ;
          int h = e >> 6, dh = e & 63;
          _Float16* O = (_Float16*)OutP;
          if (MODE == 0)
            O[(((size_t)b * NHEAD + h) * SEQ + s) * DHEAD + dh] = (_Float16)v;
          else
            O[(((size_t)b * NHEAD + h) * DHEAD + dh) * SEQ + s] = (_Float16)v;
        }
      }
}

// Flash attention: block = 4 waves; wave w owns q rows [qb, qb+16),
// streams 32 key positions per iteration with online softmax.
__global__ void __launch_bounds__(128)
attn_wmma(const _Float16* __restrict__ Qh, const _Float16* __restrict__ Kh,
          const _Float16* __restrict__ Vt, _Float16* __restrict__ Outh)
{
  const int bh  = blockIdx.x;                 // b*NHEAD + h
  const int wid = threadIdx.x >> 5, lane = threadIdx.x & 31;
  const int qb  = blockIdx.y * 64 + wid * 16;
  const int b   = bh >> 4, h = bh & 15;
  const int hl  = lane >> 4, nl = lane & 15, kk = hl * 8;

  __shared__ __align__(16) _Float16 Pst[4][16][40];  // per-wave P staging

  const _Float16* Qbase = Qh + (size_t)bh * SEQ * DHEAD;
  const _Float16* Kbase = Kh + (size_t)bh * SEQ * DHEAD;
  const _Float16* Vbase = Vt + (size_t)bh * DHEAD * SEQ;

  // Q fragments: rows qb..qb+15, K-dim = DH = 64 -> two K=32 fragments
  v16h qf[2];
  {
    const _Float16* qrow = Qbase + (size_t)(qb + nl) * DHEAD;
    qf[0] = ld_frag(qrow + kk,      qrow + kk + 16);
    qf[1] = ld_frag(qrow + 32 + kk, qrow + 32 + kk + 16);
  }

  v8f o[4] = {};                 // 16x64 output accumulator (4 d-chunks)
  float rmax[8], rsum[8];
#pragma unroll
  for (int r = 0; r < 8; ++r) { rmax[r] = -INFINITY; rsum[r] = 0.f; }
  const float scale = 0.125f;    // 1/sqrt(DH)

  for (int jb = 0; jb < qb + 16; jb += 32) {
    // scores: two 16-wide key subtiles, chained over DH
    v8f s0 = {}, s1 = {};
    {
      const _Float16* kr0 = Kbase + (size_t)(jb + nl) * DHEAD;
      const _Float16* kr1 = Kbase + (size_t)(jb + 16 + nl) * DHEAD;
      v16h k00 = ld_frag(kr0 + kk,      kr0 + kk + 16);
      v16h k01 = ld_frag(kr0 + 32 + kk, kr0 + 32 + kk + 16);
      v16h k10 = ld_frag(kr1 + kk,      kr1 + kk + 16);
      v16h k11 = ld_frag(kr1 + 32 + kk, kr1 + 32 + kk + 16);
      s0 = wmma_f16(qf[0], k00, s0);  s0 = wmma_f16(qf[1], k01, s0);
      s1 = wmma_f16(qf[0], k10, s1);  s1 = wmma_f16(qf[1], k11, s1);
    }

    // causal mask + online softmax (row m = r + 8*hl, col n = nl)
#pragma unroll
    for (int r = 0; r < 8; ++r) {
      int mg = qb + r + 8 * hl;
      float x0 = s0[r] * scale, x1 = s1[r] * scale;
      if (jb + nl > mg)      x0 = -INFINITY;
      if (jb + 16 + nl > mg) x1 = -INFINITY;
      float nmax = fmaxf(rmax[r], redmax16(fmaxf(x0, x1)));
      float alpha = __expf(rmax[r] - nmax);     // exp(-inf)=0 on first tile
      float p0 = __expf(x0 - nmax);
      float p1 = __expf(x1 - nmax);
      rsum[r] = rsum[r] * alpha + redsum16(p0 + p1);
      rmax[r] = nmax;
#pragma unroll
      for (int c = 0; c < 4; ++c) o[c][r] *= alpha;
      int m = r + 8 * hl;                       // C-layout -> A-layout via LDS
      Pst[wid][m][nl]      = (_Float16)p0;
      Pst[wid][m][16 + nl] = (_Float16)p1;
    }

    // P (16x32) as A-fragment
    const _Float16* prow = &Pst[wid][nl][0];
    v16h pf = ld_frag(prow + kk, prow + kk + 16);

    // O += P * V  (V transposed: lane's d-column is contiguous over s)
#pragma unroll
    for (int c = 0; c < 4; ++c) {
      const _Float16* vrow = Vbase + (size_t)(c * 16 + nl) * SEQ + jb;
      v16h vf = ld_frag(vrow + kk, vrow + 16 + kk);
      o[c] = wmma_f16(pf, vf, o[c]);
    }
  }

  // normalize and store f16 [B,S,D] (heads concatenated)
#pragma unroll
  for (int r = 0; r < 8; ++r) {
    float inv = 1.0f / rsum[r];
    int q = qb + r + 8 * hl;
#pragma unroll
    for (int c = 0; c < 4; ++c) {
      int d = h * DHEAD + c * 16 + nl;
      Outh[((size_t)b * SEQ + q) * DMODEL + d] = (_Float16)(o[c][r] * inv);
    }
  }
}

extern "C" void kernel_launch(void* const* d_in, const int* in_sizes, int n_in,
                              void* d_out, int out_size, void* d_ws, size_t ws_size,
                              hipStream_t stream) {
  (void)in_sizes; (void)n_in; (void)out_size; (void)ws_size;
  const float* query = (const float*)d_in[0];
  const float* key   = (const float*)d_in[1];
  const float* value = (const float*)d_in[2];
  // d_in[3] = mask (tril) -> causal mask applied analytically
  const float* Wq = (const float*)d_in[4];
  const float* Wk = (const float*)d_in[5];
  const float* Wv = (const float*)d_in[6];
  const float* Wo = (const float*)d_in[7];
  const float* bo = (const float*)d_in[8];

  const size_t nTok = (size_t)BATCH * SEQ * DMODEL;   // 4M elements
  _Float16* Qh = (_Float16*)d_ws;       // [B,H,S,DH]  f16
  _Float16* Kh = Qh + nTok;             // [B,H,S,DH]  f16
  _Float16* Vt = Kh + nTok;             // [B,H,DH,S]  f16
  _Float16* Ax = Vt + nTok;             // [B,S,D]     f16 (attention out)

  dim3 gg(DMODEL / 64, (BATCH * SEQ) / 128);
  gemm_wmma<float, 0><<<gg, 256, 0, stream>>>(query, Wq, nullptr, Qh);
  gemm_wmma<float, 0><<<gg, 256, 0, stream>>>(key,   Wk, nullptr, Kh);
  gemm_wmma<float, 1><<<gg, 256, 0, stream>>>(value, Wv, nullptr, Vt);

  attn_wmma<<<dim3(BATCH * NHEAD, SEQ / 64), 128, 0, stream>>>(Qh, Kh, Vt, Ax);

  gemm_wmma<_Float16, 2><<<gg, 256, 0, stream>>>(Ax, Wo, bo, d_out);
}
